// BertSelfAttention_59639915872737
// MI455X (gfx1250) — matrix operand order repaired and checked by reading, not compiled, not run on hardware
//
#include <hip/hip_runtime.h>
#include <hip/hip_bf16.h>

typedef __attribute__((ext_vector_type(16))) __bf16 bf16x16;
typedef __attribute__((ext_vector_type(8)))  __bf16 bf16x8;
typedef __attribute__((ext_vector_type(4)))  __bf16 bf16x4;
typedef __attribute__((ext_vector_type(8)))  float  f32x8;
typedef __attribute__((ext_vector_type(4)))  float  f32x4;
typedef __attribute__((ext_vector_type(8)))  float  v8f;
typedef __attribute__((ext_vector_type(4)))  int    i32x4;

#define BATCH 8
#define SEQ   1024
#define DMODEL 1024
#define NHEAD 16
#define HDIM  64

// ---------------------------------------------------------------------------
// Kernel 0: f32 -> bf16 conversion (hidden states and weights), 8 elems/thread.
// ---------------------------------------------------------------------------
__global__ __launch_bounds__(256) void cvt_kernel(
    const float* __restrict__ src, __bf16* __restrict__ dst, int n8)
{
    const int i = blockIdx.x * 256 + threadIdx.x;
    if (i >= n8) return;
    f32x8 v = *(const f32x8*)(src + (size_t)i * 8);
    bf16x8 o;
#pragma unroll
    for (int e = 0; e < 8; ++e) o[e] = (__bf16)v[e];
    *(bf16x8*)(dst + (size_t)i * 8) = o;
}

// ---------------------------------------------------------------------------
// Kernel 1: fused Q/K/V projection GEMM (bf16 operands, f32 accumulate).
//   M = B*S = 8192, N_total = 3*1024 (Wq|Wk|Wv concatenated in workspace).
// Each wave computes a 16x128 tile. Per k-step: one 18-load clause, one wait,
// then 8 back-to-back WMMAs, so next iteration's loads overlap the XDL pipe.
// ---------------------------------------------------------------------------
__global__ __launch_bounds__(256) void qkv_kernel(
    const __bf16* __restrict__ hsb, const __bf16* __restrict__ Wb,
    const float* __restrict__ bq, const float* __restrict__ bk,
    const float* __restrict__ bv,
    __bf16* __restrict__ Qw, __bf16* __restrict__ Kw, __bf16* __restrict__ Vtw)
{
    const int tid  = threadIdx.x;
    const int lane = tid & 31;
    const int wave = tid >> 5;
    const int gw   = blockIdx.x * 8 + wave;
    const int NT   = (3 * DMODEL) / 128;         // 24 n-tiles of width 128
    const int mt   = gw / NT;                    // 0..511
    const int nt   = gw % NT;                    // 0..23
    const int m0   = mt * 16;
    const int n0   = nt * 128;                   // never crosses a matrix bound

    const int mrow = lane & 15;                  // A row / C col within tile
    const int half = lane >> 4;                  // K-half selector

    const __bf16* arow  = hsb + (size_t)(m0 + mrow) * DMODEL;
    const __bf16* brow0 = Wb  + (size_t)(n0 + mrow) * DMODEL + half * 8;

    v8f acc[8] = {};
    for (int k0 = 0; k0 < DMODEL; k0 += 32) {
        // ---- load block: 2 A-loads + 16 B-loads, no intervening consumers
        bf16x8 a0 = *(const bf16x8*)(arow + k0 + half * 8);
        bf16x8 a1 = *(const bf16x8*)(arow + k0 + 16 + half * 8);
        bf16x16 bfr[8];
#pragma unroll
        for (int j = 0; j < 8; ++j) {
            const __bf16* bp = brow0 + (size_t)j * 16 * DMODEL + k0;
            bf16x8 b0 = *(const bf16x8*)(bp);
            bf16x8 b1 = *(const bf16x8*)(bp + 16);
#pragma unroll
            for (int e = 0; e < 8; ++e) { bfr[j][e] = b0[e]; bfr[j][e + 8] = b1[e]; }
        }
        bf16x16 af;
#pragma unroll
        for (int e = 0; e < 8; ++e) { af[e] = a0[e]; af[e + 8] = a1[e]; }
        __builtin_prefetch(arow + k0 + 64, 0, 0);   // global_prefetch_b8 ahead
        // ---- compute block: 8 consecutive WMMAs (independent accumulators)
#pragma unroll
        for (int j = 0; j < 8; ++j)
            acc[j] = __builtin_amdgcn_wmma_f32_16x16x32_bf16(
                false, af, false, bfr[j], (short)0, acc[j], false, false);
    }

    const int mat   = n0 >> 10;                  // 0=Q, 1=K, 2=V (wave-uniform)
    const int nbase = n0 & (DMODEL - 1);
    const float* bias = (mat == 0) ? bq : (mat == 1) ? bk : bv;
    const int b = m0 >> 10;
    const int sbase = m0 & (SEQ - 1);
#pragma unroll
    for (int j = 0; j < 8; ++j) {
        const int n = nbase + j * 16 + mrow;
        const int h = n >> 6;
        const int d = n & (HDIM - 1);
        const float bb = bias[n];
#pragma unroll
        for (int r = 0; r < 8; ++r) {
            const int s = sbase + 8 * half + r;
            const float val = acc[j][r] + bb;
            if (mat == 0)
                Qw[(((size_t)(b * NHEAD + h)) * SEQ + s) * HDIM + d] = (__bf16)val;
            else if (mat == 1)
                Kw[(((size_t)(b * NHEAD + h)) * SEQ + s) * HDIM + d] = (__bf16)val;
            else
                Vtw[(((size_t)(b * NHEAD + h)) * HDIM + d) * SEQ + s] = (__bf16)val;
        }
    }
}

// ---------------------------------------------------------------------------
// Kernel 2: fused attention per (b, h, 16-query tile). 4 waves / 128 threads.
//   Phase 1: scores = (Q Kt)/8 masked -> LDS f32        (WMMA, K-dim=64)
//   Phase 2: row softmax (wave32 shuffles, f32x4 lanes); probs -> global f32
//            AND -> LDS bf16 (ready-to-use WMMA A-fragments for phase 3)
//   Phase 3: context = probs @ V                        (WMMA, K-dim=1024)
// ---------------------------------------------------------------------------
#define SC_STRIDE 1032   // 1024 + pad: breaks LDS bank conflicts, keeps 16B align

__global__ __launch_bounds__(128) void attn_kernel(
    const __bf16* __restrict__ Qw, const __bf16* __restrict__ Kw,
    const __bf16* __restrict__ Vtw, const int* __restrict__ amask,
    float* __restrict__ ctx, float* __restrict__ probs)
{
    __shared__ float  sc[16][SC_STRIDE];   // f32 scores
    __shared__ __bf16 pb[16][SC_STRIDE];   // bf16 probs for phase-3 A-frags

    const int bid  = blockIdx.x;                 // B*H*(S/16) = 8192
    const int qt   = bid & 63;
    const int bh   = bid >> 6;                   // 0..127
    const int b    = bh >> 4;
    const int h    = bh & 15;
    const int m0   = qt * 16;

    const int tid  = threadIdx.x;
    const int lane = tid & 31;
    const int wave = tid >> 5;
    const int mrow = lane & 15;
    const int half = lane >> 4;

    const __bf16* Qbase = Qw  + (size_t)bh * SEQ * HDIM;
    const __bf16* Kbase = Kw  + (size_t)bh * SEQ * HDIM;
    const __bf16* Vbase = Vtw + (size_t)bh * HDIM * SEQ;
    const int*    msk   = amask + b * SEQ;

    // ---- Phase 1: 16x16 score tiles across N=1024, K-dim = 64 (2 wmma steps)
    bf16x16 qa[2];
    {
        const __bf16* qrow = Qbase + (size_t)(m0 + mrow) * HDIM;
#pragma unroll
        for (int kk = 0; kk < 2; ++kk) {
            bf16x8 lo = *(const bf16x8*)(qrow + kk * 32 + half * 8);
            bf16x8 hi = *(const bf16x8*)(qrow + kk * 32 + 16 + half * 8);
#pragma unroll
            for (int e = 0; e < 8; ++e) { qa[kk][e] = lo[e]; qa[kk][e + 8] = hi[e]; }
        }
    }
    for (int nt = wave; nt < SEQ / 16; nt += 4) {
        const int n0 = nt * 16;
        const __bf16* krow = Kbase + (size_t)(n0 + mrow) * HDIM;  // B col n = K row n
        // load block (4 x b128), then 2 chained WMMAs
        bf16x16 bfr[2];
#pragma unroll
        for (int kk = 0; kk < 2; ++kk) {
            bf16x8 lo = *(const bf16x8*)(krow + kk * 32 + half * 8);
            bf16x8 hi = *(const bf16x8*)(krow + kk * 32 + 16 + half * 8);
#pragma unroll
            for (int e = 0; e < 8; ++e) { bfr[kk][e] = lo[e]; bfr[kk][e + 8] = hi[e]; }
        }
        v8f acc = {};
#pragma unroll
        for (int kk = 0; kk < 2; ++kk)
            acc = __builtin_amdgcn_wmma_f32_16x16x32_bf16(
                false, qa[kk], false, bfr[kk], (short)0, acc, false, false);
        const int col = n0 + mrow;
        const int mk  = msk[col];
#pragma unroll
        for (int r = 0; r < 8; ++r)
            sc[8 * half + r][col] = mk ? acc[r] * 0.125f : -INFINITY;
    }
    __syncthreads();

    // ---- Phase 2: softmax; wave owns 4 rows, each lane owns 8 groups of 4 cols
    float* prow = probs + ((size_t)bh * SEQ + m0) * SEQ;
#pragma unroll 1
    for (int rr = 0; rr < 4; ++rr) {
        const int row = wave * 4 + rr;
        f32x4 v[8];
        float mx = -INFINITY;
#pragma unroll
        for (int j = 0; j < 8; ++j) {
            v[j] = *(const f32x4*)&sc[row][lane * 4 + 128 * j];
#pragma unroll
            for (int e = 0; e < 4; ++e) mx = fmaxf(mx, v[j][e]);
        }
#pragma unroll
        for (int off = 16; off; off >>= 1) mx = fmaxf(mx, __shfl_xor(mx, off, 32));
        float sum = 0.0f;
#pragma unroll
        for (int j = 0; j < 8; ++j)
#pragma unroll
            for (int e = 0; e < 4; ++e) { v[j][e] = __expf(v[j][e] - mx); sum += v[j][e]; }
#pragma unroll
        for (int off = 16; off; off >>= 1) sum += __shfl_xor(sum, off, 32);
        const float inv = 1.0f / sum;
#pragma unroll
        for (int j = 0; j < 8; ++j) {
            const int col = lane * 4 + 128 * j;
            i32x4 m4 = *(const i32x4*)(msk + col);
            f32x4 p;
            bf16x4 pb4;
#pragma unroll
            for (int e = 0; e < 4; ++e) {
                p[e]   = m4[e] ? v[j][e] * inv : 0.0f;   // second masked_fill
                pb4[e] = (__bf16)p[e];
            }
            *(f32x4*)(prow + (size_t)row * SEQ + col) = p;   // probs output (f32)
            *(bf16x4*)&pb[row][col] = pb4;                    // phase-3 operand
        }
    }
    __syncthreads();

    // ---- Phase 3: context 16x64 = probs(16x1024) @ V(1024x64); wave owns 16 cols
    {
        const int n0 = wave * 16;                                 // head-dim cols
        const __bf16* vtrow = Vbase + (size_t)(n0 + mrow) * SEQ;  // B col = Vt row
        v8f acc = {};
        for (int k0 = 0; k0 < SEQ; k0 += 64) {
            // load block: 4 ds_load_b128 + 4 global_load_b128
            bf16x16 af[2], bfr[2];
#pragma unroll
            for (int u = 0; u < 2; ++u) {
                const int k = k0 + u * 32;
                bf16x8 a0 = *(const bf16x8*)&pb[mrow][k + half * 8];
                bf16x8 a1 = *(const bf16x8*)&pb[mrow][k + 16 + half * 8];
                const __bf16* bp = vtrow + k + half * 8;
                bf16x8 b0 = *(const bf16x8*)(bp);
                bf16x8 b1 = *(const bf16x8*)(bp + 16);
#pragma unroll
                for (int e = 0; e < 8; ++e) {
                    af[u][e] = a0[e];  af[u][e + 8] = a1[e];
                    bfr[u][e] = b0[e]; bfr[u][e + 8] = b1[e];
                }
            }
            __builtin_prefetch(vtrow + k0 + 128, 0, 0);
#pragma unroll
            for (int u = 0; u < 2; ++u)
                acc = __builtin_amdgcn_wmma_f32_16x16x32_bf16(
                    false, af[u], false, bfr[u], (short)0, acc, false, false);
        }
#pragma unroll
        for (int r = 0; r < 8; ++r) {
            const int s = m0 + 8 * half + r;
            ctx[((size_t)(b * SEQ + s)) * DMODEL + h * HDIM + (n0 + mrow)] = acc[r];
        }
    }
}

// ---------------------------------------------------------------------------
extern "C" void kernel_launch(void* const* d_in, const int* in_sizes, int n_in,
                              void* d_out, int out_size, void* d_ws, size_t ws_size,
                              hipStream_t stream) {
    const float* hs  = (const float*)d_in[0];
    const int*   am  = (const int*)  d_in[1];
    const float* Wq  = (const float*)d_in[2];
    const float* bq  = (const float*)d_in[3];
    const float* Wk  = (const float*)d_in[4];
    const float* bk  = (const float*)d_in[5];
    const float* Wv  = (const float*)d_in[6];
    const float* bv  = (const float*)d_in[7];

    float* ctx   = (float*)d_out;                                  // [B,S,D]
    float* probs = ctx + (size_t)BATCH * SEQ * DMODEL;             // [B,H,S,S]

    const size_t hs_elems  = (size_t)BATCH * SEQ * DMODEL;         // 8.4M
    const size_t w_elems   = (size_t)DMODEL * DMODEL;              // 1M
    const size_t qkv_elems = (size_t)BATCH * NHEAD * SEQ * HDIM;   // 8.4M

    __bf16* hsb = (__bf16*)d_ws;               // 16 MB
    __bf16* Wb  = hsb + hs_elems;              // 6 MB (Wq|Wk|Wv)
    __bf16* Qw  = Wb + 3 * w_elems;            // 16 MB
    __bf16* Kw  = Qw + qkv_elems;              // 16 MB
    __bf16* Vtw = Kw + qkv_elems;              // 16 MB

    // f32 -> bf16 staging
    cvt_kernel<<<(int)(hs_elems / 8 / 256), 256, 0, stream>>>(hs, hsb, (int)(hs_elems / 8));
    cvt_kernel<<<(int)(w_elems / 8 / 256), 256, 0, stream>>>(Wq, Wb,               (int)(w_elems / 8));
    cvt_kernel<<<(int)(w_elems / 8 / 256), 256, 0, stream>>>(Wk, Wb + w_elems,     (int)(w_elems / 8));
    cvt_kernel<<<(int)(w_elems / 8 / 256), 256, 0, stream>>>(Wv, Wb + 2 * w_elems, (int)(w_elems / 8));

    // 512 m-tiles * 24 n-tiles = 12288 waves / 8 waves-per-block
    qkv_kernel<<<1536, 256, 0, stream>>>(hsb, Wb, bq, bk, bv, Qw, Kw, Vtw);
    // B*H*(S/16) = 8192 tiles
    attn_kernel<<<8192, 128, 0, stream>>>(Qw, Kw, Vtw, am, ctx, probs);
}
